// MultiHeadLatentAttention_24902220382972
// MI455X (gfx1250) — matrix operand order repaired and checked by reading, not compiled, not run on hardware
//
#include <hip/hip_runtime.h>

#define DEVINL __device__ __forceinline__

#if defined(__has_builtin)
#if __has_builtin(__builtin_amdgcn_tensor_load_to_lds)
#define HAVE_TDM 1
#endif
#endif

// ---------------- problem constants ----------------
constexpr int H = 16, DN = 128, DR = 64, DV = 128;
constexpr int QL = 1536, KL = 512, HID = 2048;
constexpr int Bc = 2, Sc = 1024;
constexpr int BS  = Bc * Sc;        // 2048 rows in every GEMM
constexpr int DH  = DN + DR;        // 192  (per-head q/k dim)
constexpr int DQ  = H * DH;         // 3072
constexpr int DKV = H * (DN + DV);  // 4096
constexpr int KVA = KL + DR;        // 576
constexpr int KCH = 64;             // K-chunk staged per LDS buffer

typedef __bf16 v16bf __attribute__((ext_vector_type(16)));
typedef float  v8f   __attribute__((ext_vector_type(8)));
typedef unsigned int u32x4 __attribute__((ext_vector_type(4)));
typedef int          i32x8 __attribute__((ext_vector_type(8)));
typedef int          i32x4 __attribute__((ext_vector_type(4)));

union U16bf { v16bf v; uint4 u[2]; unsigned w[8]; };

// RA/RB = operand-reuse hints (ISA 7.12 OPSEL): legal when the current WMMA is
// identical to the previous one and shares that operand's registers.
template <bool RA, bool RB>
DEVINL v8f wmma_bf16(v16bf a, v16bf b, v8f c) {
  return __builtin_amdgcn_wmma_f32_16x16x32_bf16(false, a, false, b, (short)0, c,
                                                 RA, RB);
}

// A tile 16(M)x32(K) from row-major [M][K] bf16 (stride lda elements).
DEVINL v16bf load_a_tile(const __bf16* A, int lda, int kb, int lane) {
  const int row = lane & 15;
  const int kk  = kb + ((lane >> 4) << 4);
  const __bf16* p = A + (size_t)row * lda + kk;
  U16bf t;
  t.u[0] = *(const uint4*)(p);        // K = kk .. kk+7
  t.u[1] = *(const uint4*)(p + 8);    // K = kk+8 .. kk+15
  return t.v;
}

// B tile 32(K)x16(N) from K-major storage Bt[N][K] (stride ldb) in global mem.
DEVINL v16bf load_b_tile(const __bf16* Bt, int ldb, int kb, int lane) {
  const int n  = lane & 15;
  const int kk = kb + ((lane >> 4) << 4);
  const __bf16* p = Bt + (size_t)n * ldb + kk;
  U16bf t;
  t.u[0] = *(const uint4*)(p);
  t.u[1] = *(const uint4*)(p + 8);
  return t.v;
}

// B tile from an LDS-staged panel laid out [64 n][KCH k]
DEVINL v16bf lds_b_tile(const __bf16* Lb, int nt, int kb, int lane) {
  const int n  = lane & 15;
  const int kk = kb + ((lane >> 4) << 4);
  const __bf16* p = Lb + (nt + n) * KCH + kk;
  U16bf t;
  t.u[0] = *(const uint4*)(p);
  t.u[1] = *(const uint4*)(p + 8);
  return t.v;
}

DEVINL unsigned pack2(float a, float b) {
  union { __bf16 h[2]; unsigned u; } t;
  t.h[0] = (__bf16)a; t.h[1] = (__bf16)b;
  return t.u;
}

#ifdef HAVE_TDM
// Issue one Tensor-Data-Mover DMA: 2D tile (rows x rowlen bf16) from global
// (row stride = stride_elems) into LDS at byte offset lds_off.  D# per ISA 8.3/8.4.
// This toolchain exposes the 6-arg builtin (g0, g1, g2, g3, g1-alt, cpol).
DEVINL void tdm_stage_2d(const __bf16* gsrc, unsigned lds_off,
                         int rows, int rowlen_elems, int stride_elems) {
  const unsigned long long ga = (unsigned long long)gsrc;
  u32x4 g0;
  g0[0] = 1u;                                          // count=1 valid descriptor
  g0[1] = lds_off;                                     // lds_addr (bytes)
  g0[2] = (unsigned)(ga & 0xffffffffu);                // global_addr[31:0]
  g0[3] = (unsigned)((ga >> 32) & 0x1ffffffu) | (2u << 30); // addr[56:32] | type=2
  const unsigned dim0 = (unsigned)stride_elems;        // tensor x-extent (elems)
  const unsigned dim1 = (unsigned)rows;                // tensor y-extent
  i32x8 g1;
  g1[0] = (int)(1u << 16);                             // data_size=1 (2B), mask=0
  g1[1] = (int)((dim0 & 0xffffu) << 16);               // tensor_dim0[15:0] @bit48
  g1[2] = (int)(((dim0 >> 16) & 0xffffu) | ((dim1 & 0xffffu) << 16)); // dim0 hi|dim1 lo
  g1[3] = (int)(((dim1 >> 16) & 0xffffu) |
                (((unsigned)rowlen_elems & 0xffffu) << 16));          // tile_dim0 @112
  g1[4] = (int)((unsigned)rows & 0xffffu);             // tile_dim1 @128, tile_dim2=0
  g1[5] = (int)(unsigned)stride_elems;                 // tensor_dim0_stride[31:0]
  g1[6] = 0;                                           // stride[47:32]=0, dim1_stride lo
  g1[7] = 0;
  i32x4 z4 = {0, 0, 0, 0};
  i32x8 z8 = {0, 0, 0, 0, 0, 0, 0, 0};
  __builtin_amdgcn_tensor_load_to_lds(g0, g1, z4, z4, z8, 0);
}
#endif

// fallback: whole-block cooperative copy of the 64 x KCH panel into LDS
DEVINL void fb_stage_2d(__bf16* Lb, const __bf16* Bt, int n0, int kb, int K, int tid) {
  for (int i = tid; i < 64 * KCH / 8; i += 128) {
    const int row = i >> 3, off = (i & 7) * 8;
    *(uint4*)(Lb + row * KCH + off) =
        *(const uint4*)(Bt + (size_t)(n0 + row) * K + kb + off);
  }
}

// ---------------- bf16 GEMM: C(f32,[M][N]) = A(bf16,[M][K]) * Bt(bf16,[N][K])^T
// Block = 4 waves = 64(M) x 64(N) tile.  B panel is staged once per block into
// double-buffered LDS by the TDM (async, TENSORcnt) and shared by all 4 waves;
// A tiles are register-prefetched one K-chunk ahead.  B tiles are read from
// LDS into two full register sets so the 16 ds_loads issue as one clause and
// the 8 WMMAs run back-to-back (A-operand reuse hints on WMMAs 2..4 of a set).
__global__ __launch_bounds__(128)
void gemm_bf16_wmma(const __bf16* __restrict__ A, const __bf16* __restrict__ Bt,
                    float* __restrict__ C, int M, int N, int K) {
  __shared__ alignas(16) __bf16 ldsB[2][64 * KCH];
  const int lane = threadIdx.x & 31;
  const int wave = threadIdx.x >> 5;
  const int m0 = blockIdx.y * 64 + wave * 16;
  const int n0 = blockIdx.x * 64;
  const int nchunks = K / KCH;

  // stage chunk 0 into buffer 0
#ifdef HAVE_TDM
  if (wave == 0) {
    tdm_stage_2d(Bt + (size_t)n0 * K, (unsigned)(size_t)&ldsB[0][0], 64, KCH, K);
    __builtin_amdgcn_s_wait_tensorcnt(0);
  }
#else
  fb_stage_2d(&ldsB[0][0], Bt, n0, 0, K, threadIdx.x);
#endif
  __syncthreads();

  v8f acc[4] = {};
  const __bf16* Arow = A + (size_t)m0 * K;
  v16bf a0 = load_a_tile(Arow, K, 0, lane);
  v16bf a1 = load_a_tile(Arow, K, 32, lane);

  for (int c = 0; c < nchunks; ++c) {
    const int buf = c & 1;
    // kick off async stage of chunk c+1 into the other buffer
#ifdef HAVE_TDM
    if (wave == 0 && c + 1 < nchunks)
      tdm_stage_2d(Bt + (size_t)n0 * K + (c + 1) * KCH,
                   (unsigned)(size_t)&ldsB[buf ^ 1][0], 64, KCH, K);
#else
    if (c + 1 < nchunks)
      fb_stage_2d(&ldsB[buf ^ 1][0], Bt, n0, (c + 1) * KCH, K, threadIdx.x);
#endif
    // register-prefetch next chunk's A tiles
    v16bf na0 = a0, na1 = a1;
    if (c + 1 < nchunks) {
      na0 = load_a_tile(Arow, K, (c + 1) * KCH, lane);
      na1 = load_a_tile(Arow, K, (c + 1) * KCH + 32, lane);
    }
    const __bf16* Lb = &ldsB[buf][0];
    v16bf b0[4], b1[4];
#pragma unroll
    for (int t = 0; t < 4; ++t) b0[t] = lds_b_tile(Lb, t * 16, 0, lane);
#pragma unroll
    for (int t = 0; t < 4; ++t) b1[t] = lds_b_tile(Lb, t * 16, 32, lane);
    acc[0] = wmma_bf16<false, false>(a0, b0[0], acc[0]);
    acc[1] = wmma_bf16<true,  false>(a0, b0[1], acc[1]);
    acc[2] = wmma_bf16<true,  false>(a0, b0[2], acc[2]);
    acc[3] = wmma_bf16<true,  false>(a0, b0[3], acc[3]);
    acc[0] = wmma_bf16<false, false>(a1, b1[0], acc[0]);
    acc[1] = wmma_bf16<true,  false>(a1, b1[1], acc[1]);
    acc[2] = wmma_bf16<true,  false>(a1, b1[2], acc[2]);
    acc[3] = wmma_bf16<true,  false>(a1, b1[3], acc[3]);
    a0 = na0; a1 = na1;
#ifdef HAVE_TDM
    if (wave == 0) __builtin_amdgcn_s_wait_tensorcnt(0);
#endif
    __syncthreads();
  }

  const int col = lane & 15, rb = (lane >> 4) * 8;
#pragma unroll
  for (int t = 0; t < 4; ++t)
#pragma unroll
    for (int j = 0; j < 8; ++j)
      C[(size_t)(m0 + rb + j) * N + (n0 + t * 16 + col)] = acc[t][j];
}

// ---------------- f32 -> bf16 helpers ----------------
__global__ void convert_f32_bf16(const float* __restrict__ in, __bf16* __restrict__ out,
                                 size_t n) {
  size_t i = (size_t)blockIdx.x * blockDim.x + threadIdx.x;
  const size_t stride = (size_t)gridDim.x * blockDim.x;
  for (; i < n; i += stride) out[i] = (__bf16)in[i];
}

// weight [K][ldin] f32 -> transposed bf16 [N][K] (K-major for WMMA B operand)
__global__ void convert_wT_bf16(const float* __restrict__ in, int ldin,
                                __bf16* __restrict__ outT, int N, int K) {
  size_t i = (size_t)blockIdx.x * blockDim.x + threadIdx.x;
  const size_t total = (size_t)N * K;
  const size_t stride = (size_t)gridDim.x * blockDim.x;
  for (; i < total; i += stride) {
    const int n = (int)(i / K), k = (int)(i % K);
    outT[i] = (__bf16)in[(size_t)k * ldin + n];
  }
}

// ---------------- layernorm (row per block) -> bf16 ----------------
DEVINL float block_sum(float v, float* sh) {
#pragma unroll
  for (int off = 16; off > 0; off >>= 1) v += __shfl_xor(v, off, 32);
  const int wid = threadIdx.x >> 5;
  if ((threadIdx.x & 31) == 0) sh[wid] = v;
  __syncthreads();
  float r = (threadIdx.x < (blockDim.x >> 5)) ? sh[threadIdx.x] : 0.f;
#pragma unroll
  for (int off = 4; off > 0; off >>= 1) r += __shfl_xor(r, off, 32);
  if (threadIdx.x == 0) sh[8] = r;
  __syncthreads();
  r = sh[8];
  __syncthreads();
  return r;
}

__global__ __launch_bounds__(256)
void layernorm_to_bf16(const float* __restrict__ X, int ldx, int D,
                       const float* __restrict__ g, const float* __restrict__ beta,
                       __bf16* __restrict__ Y, int ldy) {
  __shared__ float sh[9];
  const int row = blockIdx.x;
  const float* x = X + (size_t)row * ldx;
  float s = 0.f;
  for (int i = threadIdx.x; i < D; i += blockDim.x) s += x[i];
  s = block_sum(s, sh);
  const float mean = s / (float)D;
  float v = 0.f;
  for (int i = threadIdx.x; i < D; i += blockDim.x) {
    const float d = x[i] - mean; v += d * d;
  }
  v = block_sum(v, sh);
  const float rstd = rsqrtf(v / (float)D + 1e-5f);
  for (int i = threadIdx.x; i < D; i += blockDim.x)
    Y[(size_t)row * ldy + i] = (__bf16)((x[i] - mean) * rstd * g[i] + beta[i]);
}

// ---------------- rope on the shared k-rope slice (DR=64 -> 32 pairs) ----------
__global__ void rope_k_kernel(const float* __restrict__ kvfull,
                              const float* __restrict__ fc, const float* __restrict__ fs,
                              __bf16* __restrict__ krope) {
  const int bs = blockIdx.x;          // 0..BS-1
  const int s  = bs & (Sc - 1);
  const int i  = threadIdx.x;         // 0..31 pair index
  const float* src = kvfull + (size_t)bs * KVA + KL;
  const float xr = src[2 * i], xi = src[2 * i + 1];
  const float c = fc[s * (DR / 2) + i], sn = fs[s * (DR / 2) + i];
  krope[(size_t)bs * DR + 2 * i]     = (__bf16)(xr * c - xi * sn);
  krope[(size_t)bs * DR + 2 * i + 1] = (__bf16)(xr * sn + xi * c);
}

// ---------------- assemble q_full / k_full / V^T in bf16 ----------------
// qf,kf : [b][h][s][192] row-major per head;  vt : [b][h][d=128][s] (key-major)
__global__ __launch_bounds__(192)
void assemble_kernel(const float* __restrict__ qbig, const float* __restrict__ kvbig,
                     const __bf16* __restrict__ krope,
                     const float* __restrict__ fc, const float* __restrict__ fs,
                     __bf16* __restrict__ qf, __bf16* __restrict__ kf,
                     __bf16* __restrict__ vt) {
  const int s = blockIdx.x, h = blockIdx.y, b = blockIdx.z, d = threadIdx.x;
  const size_t bs = (size_t)b * Sc + s;

  const float* qp = qbig + bs * DQ + (size_t)h * DH;
  float qv;
  if (d < DN) {
    qv = qp[d];
  } else {
    const int r = d - DN, i = r >> 1;
    const float xr = qp[DN + 2 * i], xi = qp[DN + 2 * i + 1];
    const float c = fc[s * (DR / 2) + i], sn = fs[s * (DR / 2) + i];
    qv = (r & 1) ? (xr * sn + xi * c) : (xr * c - xi * sn);
  }
  const size_t bhs = (size_t)(b * H + h) * Sc + s;
  qf[bhs * DH + d] = (__bf16)qv;

  const float* kp = kvbig + bs * DKV + (size_t)h * (DN + DV);
  kf[bhs * DH + d] = (d < DN) ? (__bf16)kp[d] : krope[bs * DR + (d - DN)];

  if (d < DV)
    vt[((size_t)(b * H + h) * DV + d) * Sc + s] = (__bf16)kp[DN + d];
}

// ---------------- flash attention, S^T formulation ----------------
// One wave owns 16 queries of one (b,h). Scores computed as S^T = K * Q^T so
// the query index lands on the lane (N/col) axis: softmax state is lane-local.
__global__ __launch_bounds__(128)
void mla_attention(const __bf16* __restrict__ qf, const __bf16* __restrict__ kf,
                   const __bf16* __restrict__ vt, __bf16* __restrict__ aout) {
  const int lane = threadIdx.x & 31;
  const int wave = threadIdx.x >> 5;
  const int h = blockIdx.y, b = blockIdx.z;
  const int q0 = (blockIdx.x * 4 + wave) * 16;

  const __bf16* Q  = qf + ((size_t)(b * H + h) * Sc + q0) * DH;
  const __bf16* Kp = kf + (size_t)(b * H + h) * Sc * DH;
  const __bf16* Vt = vt + (size_t)(b * H + h) * DV * Sc;

  // Q^T as WMMA B operand: Q rows are K-major already ([query][d]); 6 d-chunks.
  v16bf Qb[6];
#pragma unroll
  for (int c = 0; c < 6; ++c) Qb[c] = load_b_tile(Q, DH, c * 32, lane);

  v8f O[8] = {};                       // O^T tiles: [d-tile 0..7][16x16]
  float m = -__builtin_inff(), lsum = 0.f;
  const int col = lane & 15, rb = (lane >> 4) * 8;
  const float scale = 0.0721687836487032f;  // 1/sqrt(192)
  const int qabs = q0 + col;
  const int npan = (q0 + 15) / 32 + 1;      // causal: key panels up to diagonal

  for (int kp = 0; kp < npan; ++kp) {
    // ---- S^T panel: 32 keys (two M-tiles) x 16 queries, K-dim = 192 ----
    // batch two d-chunks of K-tiles per wait; s2's WMMA reuses s1's B operand.
    v8f s1 = {}, s2 = {};
    const __bf16* K0 = Kp + (size_t)(kp * 32) * DH;
#pragma unroll
    for (int c = 0; c < 6; c += 2) {
      v16bf ka0 = load_a_tile(K0, DH, c * 32, lane);
      v16bf ka1 = load_a_tile(K0 + (size_t)16 * DH, DH, c * 32, lane);
      v16bf ka2 = load_a_tile(K0, DH, c * 32 + 32, lane);
      v16bf ka3 = load_a_tile(K0 + (size_t)16 * DH, DH, c * 32 + 32, lane);
      s1 = wmma_bf16<false, false>(ka0, Qb[c], s1);
      s2 = wmma_bf16<false, true >(ka1, Qb[c], s2);
      s1 = wmma_bf16<false, false>(ka2, Qb[c + 1], s1);
      s2 = wmma_bf16<false, true >(ka3, Qb[c + 1], s2);
    }
    // keys held by this lane: s1[j] -> kp*32 + rb + j ; s2[j] -> + 16
    float p1[8], p2[8];
    float mx = -__builtin_inff();
#pragma unroll
    for (int j = 0; j < 8; ++j) {
      const int k1 = kp * 32 + rb + j;
      const float v1 = (k1      <= qabs) ? s1[j] * scale : -__builtin_inff();
      const float v2 = (k1 + 16 <= qabs) ? s2[j] * scale : -__builtin_inff();
      p1[j] = v1; p2[j] = v2;
      mx = fmaxf(mx, fmaxf(v1, v2));
    }
    mx = fmaxf(mx, __shfl_xor(mx, 16, 32));   // other half of this query's keys
    const float mnew = fmaxf(m, mx);
    const float corr = __expf(m - mnew);
    float ps = 0.f;
#pragma unroll
    for (int j = 0; j < 8; ++j) {
      p1[j] = __expf(p1[j] - mnew);
      p2[j] = __expf(p2[j] - mnew);
      ps += p1[j] + p2[j];
    }
    ps += __shfl_xor(ps, 16, 32);
    lsum = lsum * corr + ps;
    m = mnew;
#pragma unroll
    for (int t = 0; t < 8; ++t)
#pragma unroll
      for (int j = 0; j < 8; ++j) O[t][j] *= corr;

    // ---- rearrange P^T (C layout) -> B operand layout (K=32 keys) ----
    // B vgpr j: lane<16 keys 2j,2j+1 ; lane>=16 keys 16+2j,17+2j.
    U16bf P;
#pragma unroll
    for (int j = 0; j < 4; ++j) {
      const unsigned lo1 = pack2(p1[2 * j], p1[2 * j + 1]);
      const unsigned lo2 = pack2(p2[2 * j], p2[2 * j + 1]);
      const unsigned o1 = (unsigned)__shfl_xor((int)lo1, 16, 32);
      const unsigned o2 = (unsigned)__shfl_xor((int)lo2, 16, 32);
      P.w[j]     = (lane < 16) ? lo1 : o2;
      P.w[j + 4] = (lane < 16) ? o1  : lo2;
    }
    // ---- O^T += V^T(16d x 32keys) * P^T(32keys x 16q), per d-tile ----
    // batch 4 V tiles per wait; all PV WMMAs share P as B operand (RB reuse).
#pragma unroll
    for (int t = 0; t < 8; t += 4) {
      v16bf av0 = load_a_tile(Vt + (size_t)((t + 0) * 16) * Sc, Sc, kp * 32, lane);
      v16bf av1 = load_a_tile(Vt + (size_t)((t + 1) * 16) * Sc, Sc, kp * 32, lane);
      v16bf av2 = load_a_tile(Vt + (size_t)((t + 2) * 16) * Sc, Sc, kp * 32, lane);
      v16bf av3 = load_a_tile(Vt + (size_t)((t + 3) * 16) * Sc, Sc, kp * 32, lane);
      O[t + 0] = (t == 0) ? wmma_bf16<false, false>(av0, P.v, O[t + 0])
                          : wmma_bf16<false, true >(av0, P.v, O[t + 0]);
      O[t + 1] = wmma_bf16<false, true>(av1, P.v, O[t + 1]);
      O[t + 2] = wmma_bf16<false, true>(av2, P.v, O[t + 2]);
      O[t + 3] = wmma_bf16<false, true>(av3, P.v, O[t + 3]);
    }
  }

  // epilogue: O^T lane holds (q = q0+col, d rows t*16+rb..+7) -> contiguous store
  const float inv = 1.f / lsum;
  __bf16* outp = aout + ((size_t)b * Sc + q0 + col) * (size_t)(H * DV) + h * DV;
#pragma unroll
  for (int t = 0; t < 8; ++t) {
    union { __bf16 h8[8]; uint4 u; } pk;
#pragma unroll
    for (int j = 0; j < 8; ++j) pk.h8[j] = (__bf16)(O[t][j] * inv);
    *(uint4*)(outp + t * 16 + rb) = pk.u;
  }
}

// ---------------- host-side orchestration ----------------
extern "C" void kernel_launch(void* const* d_in, const int* in_sizes, int n_in,
                              void* d_out, int out_size, void* d_ws, size_t ws_size,
                              hipStream_t stream) {
  (void)in_sizes; (void)n_in; (void)out_size; (void)ws_size;
  const float* x       = (const float*)d_in[0];
  const float* w_qkv_a = (const float*)d_in[1];
  const float* q_ln_g  = (const float*)d_in[2];
  const float* q_ln_b  = (const float*)d_in[3];
  const float* w_q_b   = (const float*)d_in[4];
  const float* w_kv_a  = (const float*)d_in[5];
  const float* kv_ln_g = (const float*)d_in[6];
  const float* kv_ln_b = (const float*)d_in[7];
  const float* w_kv_b  = (const float*)d_in[8];
  const float* w_o     = (const float*)d_in[9];
  const float* fcos    = (const float*)d_in[10];
  const float* fsin    = (const float*)d_in[11];

  char* wptr = (char*)d_ws;
  auto carve = [&](size_t bytes) {
    char* p = wptr; wptr += (bytes + 255) & ~(size_t)255; return (void*)p;
  };
  __bf16* xb    = (__bf16*)carve((size_t)BS * HID * 2);
  __bf16* wtqkv = (__bf16*)carve((size_t)QL * HID * 2);   // [QL][HID]
  __bf16* wtqb  = (__bf16*)carve((size_t)DQ * QL * 2);    // [DQ][QL]
  __bf16* wtkva = (__bf16*)carve((size_t)KVA * HID * 2);  // [KVA][HID]
  __bf16* wtkvb = (__bf16*)carve((size_t)DKV * KL * 2);   // [DKV][KL]
  __bf16* wto   = (__bf16*)carve((size_t)HID * HID * 2);  // [HID][HID]
  float*  qkv   = (float*) carve((size_t)BS * QL * 4);
  float*  kvfl  = (float*) carve((size_t)BS * KVA * 4);
  __bf16* qln   = (__bf16*)carve((size_t)BS * QL * 2);
  __bf16* kvln  = (__bf16*)carve((size_t)BS * KL * 2);
  float*  qbig  = (float*) carve((size_t)BS * DQ * 4);
  float*  kvbig = (float*) carve((size_t)BS * DKV * 4);
  __bf16* krope = (__bf16*)carve((size_t)BS * DR * 2);
  __bf16* qf    = (__bf16*)carve((size_t)Bc * H * Sc * DH * 2);
  __bf16* kf    = (__bf16*)carve((size_t)Bc * H * Sc * DH * 2);
  __bf16* vt    = (__bf16*)carve((size_t)Bc * H * DV * Sc * 2);
  __bf16* aout  = (__bf16*)carve((size_t)BS * H * DV * 2);

  const int CT = 256;
  // bf16 conversion / K-major weight transposes (per-launch, deterministic)
  convert_f32_bf16<<<4096, CT, 0, stream>>>(x, xb, (size_t)BS * HID);
  convert_wT_bf16<<<4096, CT, 0, stream>>>(w_qkv_a, QL + KL + DR, wtqkv, QL, HID);
  convert_wT_bf16<<<4096, CT, 0, stream>>>(w_q_b, DQ, wtqb, DQ, QL);
  convert_wT_bf16<<<4096, CT, 0, stream>>>(w_kv_a, KVA, wtkva, KVA, HID);
  convert_wT_bf16<<<4096, CT, 0, stream>>>(w_kv_b, DKV, wtkvb, DKV, KL);
  convert_wT_bf16<<<4096, CT, 0, stream>>>(w_o, HID, wto, HID, HID);

  // projections (only the QL columns of w_qkv_a are ever consumed)
  gemm_bf16_wmma<<<dim3(QL / 64, BS / 64), 128, 0, stream>>>(xb, wtqkv, qkv, BS, QL, HID);
  gemm_bf16_wmma<<<dim3(KVA / 64, BS / 64), 128, 0, stream>>>(xb, wtkva, kvfl, BS, KVA, HID);

  layernorm_to_bf16<<<BS, 256, 0, stream>>>(qkv, QL, QL, q_ln_g, q_ln_b, qln, QL);
  layernorm_to_bf16<<<BS, 256, 0, stream>>>(kvfl, KVA, KL, kv_ln_g, kv_ln_b, kvln, KL);
  rope_k_kernel<<<BS, 32, 0, stream>>>(kvfl, fcos, fsin, krope);

  gemm_bf16_wmma<<<dim3(DQ / 64, BS / 64), 128, 0, stream>>>(qln, wtqb, qbig, BS, DQ, QL);
  gemm_bf16_wmma<<<dim3(DKV / 64, BS / 64), 128, 0, stream>>>(kvln, wtkvb, kvbig, BS, DKV, KL);

  assemble_kernel<<<dim3(Sc, H, Bc), DH, 0, stream>>>(qbig, kvbig, krope, fcos, fsin,
                                                      qf, kf, vt);

  mla_attention<<<dim3(Sc / 64, H, Bc), 128, 0, stream>>>(qf, kf, vt, aout);

  // output projection straight into d_out (f32)
  gemm_bf16_wmma<<<dim3(HID / 64, BS / 64), 128, 0, stream>>>(aout, wto, (float*)d_out,
                                                              BS, HID, HID);
}
